// SimpleSeq2SeqRNN_11416023073121
// MI455X (gfx1250) — compile-verified, matrix operand order checked by reading, not run on hardware
//
#include <hip/hip_runtime.h>
#include <stdint.h>

// ---------------------------------------------------------------------------
// CDNA5 (gfx1250, wave32) WMMA types
// ---------------------------------------------------------------------------
typedef __attribute__((ext_vector_type(16))) __bf16 v16bf;
typedef __attribute__((ext_vector_type(8)))  float  v8f;

#define SEQ    128
#define BATCH  64
#define EMB    1024
#define HID    1024
#define NOUT   32000
#define MROWS  (SEQ * BATCH)   // 8192

// FC GEMM LDS tiling
#define BK     32              // K-slab per stage
#define PAD    40              // LDS row stride in bf16 elems (80B -> conflict-free)
#define KT     (HID / BK)      // 32 K-steps

// ---------------------------------------------------------------------------
// Helpers
// ---------------------------------------------------------------------------
static __device__ __forceinline__ uint16_t bf16_bits(float f) {
    union { __bf16 b; uint16_t u; } cv;
    cv.b = (__bf16)f;
    return cv.u;
}

// 16-bit A/B fragment (16x32 / 32x16), per ISA 7.12.2 layout:
// lanes 0-15 : row/col = lane,    K = k0+0..7   and k0+16..23
// lanes 16-31: row/col = lane-16, K = k0+8..15  and k0+24..31
static __device__ __forceinline__ v16bf ld_frag_bf16(const uint16_t* __restrict__ row,
                                                     int k0, int half) {
    union { uint4 u[2]; v16bf v; } f;
    f.u[0] = *reinterpret_cast<const uint4*>(row + k0 + half * 8);
    f.u[1] = *reinterpret_cast<const uint4*>(row + k0 + 16 + half * 8);
    return f.v;
}

// Fragment read from an LDS-resident K-slab row (32 valid elems, stride PAD)
static __device__ __forceinline__ v16bf ld_frag_lds(const uint16_t* row, int half) {
    union { uint4 u[2]; v16bf v; } f;
    f.u[0] = *reinterpret_cast<const uint4*>(row + half * 8);
    f.u[1] = *reinterpret_cast<const uint4*>(row + 16 + half * 8);
    return f.v;
}

// Same fragment, sourced from f32 memory with in-register bf16 convert
static __device__ __forceinline__ v16bf ld_frag_f32(const float* __restrict__ row,
                                                    int k0, int half) {
    const float4 a = *reinterpret_cast<const float4*>(row + k0 + half * 8);
    const float4 b = *reinterpret_cast<const float4*>(row + k0 + half * 8 + 4);
    const float4 c = *reinterpret_cast<const float4*>(row + k0 + 16 + half * 8);
    const float4 d = *reinterpret_cast<const float4*>(row + k0 + 16 + half * 8 + 4);
    v16bf v;
    v[0]  = (__bf16)a.x; v[1]  = (__bf16)a.y; v[2]  = (__bf16)a.z; v[3]  = (__bf16)a.w;
    v[4]  = (__bf16)b.x; v[5]  = (__bf16)b.y; v[6]  = (__bf16)b.z; v[7]  = (__bf16)b.w;
    v[8]  = (__bf16)c.x; v[9]  = (__bf16)c.y; v[10] = (__bf16)c.z; v[11] = (__bf16)c.w;
    v[12] = (__bf16)d.x; v[13] = (__bf16)d.y; v[14] = (__bf16)d.z; v[15] = (__bf16)d.w;
    return v;
}

static __device__ __forceinline__ v8f wmma_bf16(v16bf a, v16bf b, v8f c) {
    // (neg_a, A, neg_b, B, c_mod, C, reuse_a, reuse_b)
    return __builtin_amdgcn_wmma_f32_16x16x32_bf16(false, a, false, b, (short)0, c,
                                                   false, false);
}

static __device__ __forceinline__ v8f zero_v8f() {
    v8f z = {0.f, 0.f, 0.f, 0.f, 0.f, 0.f, 0.f, 0.f};
    return z;
}

// gfx1250 async copy global -> LDS, 16B per lane, tracked by ASYNCcnt.
// LDS operand is the low 32 bits of the generic pointer (flat->LDS truncation,
// ISA 10.2 aperture rules). Inline asm keeps this portable across toolchains.
static __device__ __forceinline__ void async_ld_b128(const uint16_t* gptr,
                                                     uint16_t* lptr) {
    uint32_t lds_off = (uint32_t)(uintptr_t)lptr;
    asm volatile("global_load_async_to_lds_b128 %0, %1, off"
                 :: "v"(lds_off), "v"(gptr) : "memory");
}

// ---------------------------------------------------------------------------
// Prep kernels
// ---------------------------------------------------------------------------
__global__ void k_f32_to_bf16(const float* __restrict__ src,
                              uint16_t* __restrict__ dst, int n) {
    int i = blockIdx.x * blockDim.x + threadIdx.x;
    int stride = gridDim.x * blockDim.x;
    for (; i < n; i += stride) dst[i] = bf16_bits(src[i]);
}

__global__ void k_copy_f32(const float* __restrict__ src,
                           float* __restrict__ dst, int n) {
    int i = blockIdx.x * blockDim.x + threadIdx.x;
    int stride = gridDim.x * blockDim.x;
    for (; i < n; i += stride) dst[i] = src[i];
}

// ---------------------------------------------------------------------------
// Kernel 2: xh = gather(emb, x) @ W_ih^T + b_ih + b_hh
// block = 256 thr (8 waves), block tile 64(M) x 256(N), wave tile 32x64
// ---------------------------------------------------------------------------
__global__ __launch_bounds__(256) void k_xh_gemm(
    const int* __restrict__ x, const float* __restrict__ emb,
    const uint16_t* __restrict__ Wih, const float* __restrict__ b_ih,
    const float* __restrict__ b_hh, float* __restrict__ xh) {
    const int lane = threadIdx.x & 31;
    const int w    = threadIdx.x >> 5;
    const int half = lane >> 4;
    const int lr   = lane & 15;
    const int m0 = blockIdx.y * 64  + (w >> 2) * 32;
    const int n0 = blockIdx.x * 256 + (w & 3)  * 64;

    const float* arow[2];
#pragma unroll
    for (int mt = 0; mt < 2; ++mt) {
        int row = m0 + mt * 16 + lr;
        arow[mt] = emb + (size_t)x[row] * EMB;   // fused embedding gather
    }
    const uint16_t* brow[4];
#pragma unroll
    for (int nt = 0; nt < 4; ++nt)
        brow[nt] = Wih + (size_t)(n0 + nt * 16 + lr) * EMB;

    v8f acc[2][4];
#pragma unroll
    for (int mt = 0; mt < 2; ++mt)
#pragma unroll
        for (int nt = 0; nt < 4; ++nt) acc[mt][nt] = zero_v8f();

    for (int k0 = 0; k0 < EMB; k0 += 32) {
        v16bf a[2], b[4];
#pragma unroll
        for (int mt = 0; mt < 2; ++mt) a[mt] = ld_frag_f32(arow[mt], k0, half);
#pragma unroll
        for (int nt = 0; nt < 4; ++nt) b[nt] = ld_frag_bf16(brow[nt], k0, half);
#pragma unroll
        for (int mt = 0; mt < 2; ++mt)
#pragma unroll
            for (int nt = 0; nt < 4; ++nt)
                acc[mt][nt] = wmma_bf16(a[mt], b[nt], acc[mt][nt]);
    }

#pragma unroll
    for (int mt = 0; mt < 2; ++mt)
#pragma unroll
        for (int nt = 0; nt < 4; ++nt) {
            int col = n0 + nt * 16 + lr;
            float bias = b_ih[col] + b_hh[col];
#pragma unroll
            for (int v = 0; v < 8; ++v) {
                int rowm = m0 + mt * 16 + half * 8 + v;
                xh[(size_t)rowm * HID + col] = acc[mt][nt][v] + bias;
            }
        }
}

// ---------------------------------------------------------------------------
// Kernel 3: one RNN step  h_new = tanh(xh_t + h_prev @ W_hh^T)
// M=64, N=1024, K=1024.  16 blocks x 8 waves, wave tile 16x32.
// ---------------------------------------------------------------------------
__global__ __launch_bounds__(256) void k_rnn_step(
    const float* __restrict__ hprev, const float* __restrict__ xh_t,
    const uint16_t* __restrict__ Whh, float* __restrict__ hnew,
    uint16_t* __restrict__ hall_t) {
    const int lane = threadIdx.x & 31;
    const int w    = threadIdx.x >> 5;
    const int half = lane >> 4;
    const int lr   = lane & 15;
    const int m0 = (w >> 1) * 16;
    const int n0 = blockIdx.x * 64 + (w & 1) * 32;

    const float* arow = hprev + (size_t)(m0 + lr) * HID;
    const uint16_t* brow0 = Whh + (size_t)(n0 + lr) * HID;
    const uint16_t* brow1 = Whh + (size_t)(n0 + 16 + lr) * HID;

    v8f acc0 = zero_v8f(), acc1 = zero_v8f();
    for (int k0 = 0; k0 < HID; k0 += 32) {
        v16bf a  = ld_frag_f32(arow, k0, half);
        v16bf b0 = ld_frag_bf16(brow0, k0, half);
        v16bf b1 = ld_frag_bf16(brow1, k0, half);
        acc0 = wmma_bf16(a, b0, acc0);
        acc1 = wmma_bf16(a, b1, acc1);
    }

#pragma unroll
    for (int nt = 0; nt < 2; ++nt) {
        const v8f acc = nt ? acc1 : acc0;
        int col = n0 + nt * 16 + lr;
#pragma unroll
        for (int v = 0; v < 8; ++v) {
            int row = m0 + half * 8 + v;
            size_t idx = (size_t)row * HID + col;
            float val = tanhf(acc[v] + xh_t[idx]);
            hnew[idx]   = val;
            hall_t[idx] = bf16_bits(val);
        }
    }
}

// ---------------------------------------------------------------------------
// Kernel 4 (dominant): out = H_all @ W_fc^T + b_fc  (8192 x 32000, K=1024)
// Block tile 128x128, 8 waves (4 M-groups x 2 N-groups), wave tile 32x64.
// Double-buffered LDS K-slabs filled with global_load_async_to_lds_b128
// (ASYNCcnt), fragments read back with ds_load_b128 (conflict-free PAD=40).
// ---------------------------------------------------------------------------
__global__ __launch_bounds__(256) void k_fc_gemm(
    const uint16_t* __restrict__ Hall, const uint16_t* __restrict__ Wfc,
    const float* __restrict__ b_fc, float* __restrict__ out) {
    __shared__ __align__(16) uint16_t smA[2][128 * PAD];  // 2 x 10 KB
    __shared__ __align__(16) uint16_t smB[2][128 * PAD];  // 2 x 10 KB

    const int tid  = threadIdx.x;
    const int lane = tid & 31;
    const int w    = tid >> 5;
    const int half = lane >> 4;
    const int lr   = lane & 15;
    const int mg   = w >> 1;   // 0..3  -> 32-row group
    const int ng   = w & 1;    // 0..1  -> 64-col group

    const uint16_t* gA = Hall + (size_t)(blockIdx.y * 128) * HID;
    const uint16_t* gB = Wfc  + (size_t)(blockIdx.x * 128) * HID;

    // Each thread moves 4 x 16B per stage: 2 chunks of A, 2 of B.
    // chunk c (0..511): row = c>>2, 16B sub-chunk = c&3 of the 64B K-slab row.
    auto issue_tile = [&](int k0, int buf) {
#pragma unroll
        for (int i = 0; i < 2; ++i) {
            int c   = tid + i * 256;
            int row = c >> 2;
            int sub = c & 3;
            async_ld_b128(gA + (size_t)row * HID + k0 + sub * 8,
                          &smA[buf][row * PAD + sub * 8]);
            async_ld_b128(gB + (size_t)row * HID + k0 + sub * 8,
                          &smB[buf][row * PAD + sub * 8]);
        }
    };

    v8f acc[2][4];
#pragma unroll
    for (int mt = 0; mt < 2; ++mt)
#pragma unroll
        for (int nt = 0; nt < 4; ++nt) acc[mt][nt] = zero_v8f();

    issue_tile(0, 0);
    for (int kt = 0; kt < KT; ++kt) {
        const int  cur  = kt & 1;
        const bool more = (kt + 1) < KT;
        if (more) {
            issue_tile((kt + 1) * BK, cur ^ 1);
            // previous tile's 4 ops done, next tile's 4 still in flight
            asm volatile("s_wait_asynccnt 0x4" ::: "memory");
        } else {
            asm volatile("s_wait_asynccnt 0x0" ::: "memory");
        }
        __syncthreads();

        v16bf a[2], b[4];
#pragma unroll
        for (int mt = 0; mt < 2; ++mt)
            a[mt] = ld_frag_lds(&smA[cur][(mg * 32 + mt * 16 + lr) * PAD], half);
#pragma unroll
        for (int nt = 0; nt < 4; ++nt)
            b[nt] = ld_frag_lds(&smB[cur][(ng * 64 + nt * 16 + lr) * PAD], half);
#pragma unroll
        for (int mt = 0; mt < 2; ++mt)
#pragma unroll
            for (int nt = 0; nt < 4; ++nt)
                acc[mt][nt] = wmma_bf16(a[mt], b[nt], acc[mt][nt]);

        if (more) __syncthreads();  // protect buf cur before next overwrite
    }

    const int m0 = blockIdx.y * 128 + mg * 32;
    const int n0 = blockIdx.x * 128 + ng * 64;
#pragma unroll
    for (int mt = 0; mt < 2; ++mt)
#pragma unroll
        for (int nt = 0; nt < 4; ++nt) {
            int col = n0 + nt * 16 + lr;
            float bias = b_fc[col];
#pragma unroll
            for (int v = 0; v < 8; ++v) {
                int rowm = m0 + mt * 16 + half * 8 + v;
                out[(size_t)rowm * NOUT + col] = acc[mt][nt][v] + bias;
            }
        }
}

// ---------------------------------------------------------------------------
// Host-side orchestration
// ---------------------------------------------------------------------------
extern "C" void kernel_launch(void* const* d_in, const int* in_sizes, int n_in,
                              void* d_out, int out_size, void* d_ws, size_t ws_size,
                              hipStream_t stream) {
    const int*   x       = (const int*)  d_in[0];  // [SEQ, BATCH]
    const float* hidden0 = (const float*)d_in[1];  // [BATCH, HID]
    const float* emb     = (const float*)d_in[2];  // [VOCAB, EMB]
    const float* W_ih    = (const float*)d_in[3];  // [HID, EMB]
    const float* W_hh    = (const float*)d_in[4];  // [HID, HID]
    const float* b_ih    = (const float*)d_in[5];  // [HID]
    const float* b_hh    = (const float*)d_in[6];  // [HID]
    const float* W_fc    = (const float*)d_in[7];  // [NOUT, HID]
    const float* b_fc    = (const float*)d_in[8];  // [NOUT]
    float* out = (float*)d_out;                    // [SEQ,BATCH,NOUT] ++ [BATCH,HID]

    char* ws = (char*)d_ws;
    size_t off = 0;
    uint16_t* Wih_bf = (uint16_t*)(ws + off); off += (size_t)HID  * EMB * 2;
    uint16_t* Whh_bf = (uint16_t*)(ws + off); off += (size_t)HID  * HID * 2;
    uint16_t* Wfc_bf = (uint16_t*)(ws + off); off += (size_t)NOUT * HID * 2;
    float*    xh     = (float*)   (ws + off); off += (size_t)MROWS * HID * 4;
    uint16_t* Hall   = (uint16_t*)(ws + off); off += (size_t)MROWS * HID * 2;
    float*    hbuf0  = (float*)   (ws + off); off += (size_t)BATCH * HID * 4;
    float*    hbuf1  = (float*)   (ws + off); off += (size_t)BATCH * HID * 4;
    (void)in_sizes; (void)n_in; (void)out_size; (void)ws_size;

    // 1) weight conversion to bf16 + hidden-state init
    k_f32_to_bf16<<<1024, 256, 0, stream>>>(W_ih, Wih_bf, HID * EMB);
    k_f32_to_bf16<<<1024, 256, 0, stream>>>(W_hh, Whh_bf, HID * HID);
    k_f32_to_bf16<<<4096, 256, 0, stream>>>(W_fc, Wfc_bf, NOUT * HID);
    k_copy_f32<<<64, 256, 0, stream>>>(hidden0, hbuf0, BATCH * HID);

    // 2) xh = embed-gather @ W_ih^T + b_ih + b_hh
    k_xh_gemm<<<dim3(EMB / 256, MROWS / 64), 256, 0, stream>>>(
        x, emb, Wih_bf, b_ih, b_hh, xh);

    // 3) sequential recurrence, stream-ordered (128 graph nodes on replay)
    for (int t = 0; t < SEQ; ++t) {
        const float* hp = (t & 1) ? hbuf1 : hbuf0;
        float*       hn = (t & 1) ? hbuf0 : hbuf1;
        k_rnn_step<<<HID / 64, 256, 0, stream>>>(
            hp, xh + (size_t)t * BATCH * HID, Whh_bf, hn,
            Hall + (size_t)t * BATCH * HID);
    }

    // 4) dominant GEMM: logits = H_all @ W_fc^T + b_fc
    k_fc_gemm<<<dim3(NOUT / 128, MROWS / 128), 256, 0, stream>>>(
        Hall, Wfc_bf, b_fc, out);

    // 5) final hidden state (after t=127 the ping-pong result is in hbuf0)
    k_copy_f32<<<64, 256, 0, stream>>>(hbuf0, out + (size_t)MROWS * NOUT,
                                       BATCH * HID);
}
